// QNetwork_45002667327552
// MI455X (gfx1250) — matrix operand order repaired and checked by reading, not compiled, not run on hardware
//
#include <hip/hip_runtime.h>

#define N_AGENTS 65536
#define N_PAIRS  1000000

typedef __attribute__((ext_vector_type(16))) _Float16 v16h;
typedef __attribute__((ext_vector_type(8)))  _Float16 v8h;
typedef __attribute__((ext_vector_type(8)))  float    v8f;

// Packed B-tile bases (in 512-half tiles). Each tile: lane L holds
// B[k=(j>>3)*16+(L>>4)*8+(j&7)][n=(L&15)] for j=0..15 (16 halfs = 32B contiguous).
#define C1T  0
#define R1AT 18
#define R1BT 36
#define R2AT 54
#define R2BT 72
#define C2T  90
#define LINT 91
#define Q1T  117
#define Q2T  121
#define Q3T  123
#define NTILES 124

// A-operand loader: WMMA 16-bit A layout — lane L holds row M=(L&15),
// halfs K = {klo..klo+7, klo+16..klo+23}, klo = (L>>4)*8.
__device__ __forceinline__ v16h load_a(const _Float16* rowbase, int klo) {
    v8h lo = *(const v8h*)(rowbase + klo);
    v8h hi = *(const v8h*)(rowbase + klo + 16);
    return __builtin_shufflevector(lo, hi, 0,1,2,3,4,5,6,7,8,9,10,11,12,13,14,15);
}

__device__ __forceinline__ v16h load_b(const _Float16* packW, int tile, int lane) {
    return *(const v16h*)(packW + (size_t)tile * 512 + lane * 16);
}

__device__ __forceinline__ float lrelu(float v) { return v > 0.f ? v : 0.01f * v; }

// ---------------------------------------------------------------------------
// Kernel 1: repack all weight matrices (f32, torch [O,K] layouts) into the
// per-lane WMMA B layout in f16. One block per tile, 512 threads = one half.
// ---------------------------------------------------------------------------
__global__ void pack_weights(const float* w_c1, const float* w_r1a, const float* w_r1b,
                             const float* w_r2a, const float* w_r2b, const float* w_c2,
                             const float* w_lin, const float* w_q1, const float* w_q2,
                             const float* w_q3, _Float16* packW) {
    int tile = blockIdx.x;
    int t = threadIdx.x;          // 0..511
    int L = t >> 4, j = t & 15;
    int kk = ((j >> 3) << 4) + ((L >> 4) << 3) + (j & 7); // k within 32-ktile
    int nn = L & 15;              // n within 16-ntile
    float v = 0.f;
    if (tile < 18) {                         // conv1: 9 taps x 2 ntiles, Cin=7 padded to 32
        int lt = tile, tap = lt >> 1, nt = lt & 1;
        int ky = tap / 3, kx = tap % 3, o = nt * 16 + nn;
        if (kk < 7) v = w_c1[((o * 7 + kk) * 3 + ky) * 3 + kx];
    } else if (tile < 90) {                  // 4 residual convs: 9 taps x 2 ntiles, Cin=32
        int m = (tile - 18) / 18, lt = (tile - 18) % 18;
        const float* w = (m == 0) ? w_r1a : (m == 1) ? w_r1b : (m == 2) ? w_r2a : w_r2b;
        int tap = lt >> 1, nt = lt & 1;
        int ky = tap / 3, kx = tap % 3, o = nt * 16 + nn;
        v = w[((o * 32 + kk) * 3 + ky) * 3 + kx];
    } else if (tile == 90) {                 // 1x1 conv 32->16
        v = w_c2[nn * 32 + kk];
    } else if (tile < 117) {                 // linear 400(->416) x 32
        int lt = tile - 91, kt = lt >> 1, nt = lt & 1;
        int k = kt * 32 + kk, o = nt * 16 + nn;
        if (k < 400) v = w_lin[o * 400 + k];
    } else if (tile < 121) {                 // q1: 64 x 32
        int lt = tile - 117, kt = lt >> 1, nt = lt & 1;
        v = w_q1[(nt * 16 + nn) * 64 + kt * 32 + kk];
    } else if (tile < 123) {                 // q2: 32 x 32
        int nt = tile - 121;
        v = w_q2[(nt * 16 + nn) * 32 + kk];
    } else {                                 // q3: 32 x 2 (padded to 16 cols)
        if (nn < 2) v = w_q3[nn * 32 + kk];
    }
    packW[(size_t)tile * 512 + L * 16 + j] = (_Float16)v;
}

// ---------------------------------------------------------------------------
// Kernel 2: fused encoder. 256 threads = 8 waves = 8 agents per workgroup.
// Whole conv stack lives in LDS; only enc[N,32] goes to HBM.
// ---------------------------------------------------------------------------
__device__ __forceinline__ void conv3x3_same(const _Float16 (*src)[32], _Float16 (*dst)[32],
                                             const _Float16* packW, int tileBase,
                                             const float* bias, const _Float16 (*resid)[32],
                                             int lane) {
    const int klo = (lane >> 4) * 8;
    const int nsel = lane & 15;
    const int msel = lane & 15;
    const int hi8 = (lane >> 4) << 3;
    for (int nt = 0; nt < 2; ++nt) {
        float bv = bias[nt * 16 + nsel];
        for (int mt = 0; mt < 2; ++mt) {
            v8f acc = {bv, bv, bv, bv, bv, bv, bv, bv};
            int p = mt * 16 + msel;
            int py = p / 5, px = p % 5;
            for (int tap = 0; tap < 9; ++tap) {
                int dy = tap / 3 - 1, dx = tap % 3 - 1;
                int iy = py + dy, ix = px + dx;
                bool ok = (p < 25) && (iy >= 0) && (iy < 5) && (ix >= 0) && (ix < 5);
                int rin = ok ? (iy * 5 + ix) : 25;   // row 25 = zero row
                v16h a = load_a(&src[rin][0], klo);
                v16h b = load_b(packW, tileBase + tap * 2 + nt, lane);
                acc = __builtin_amdgcn_wmma_f32_16x16x32_f16(false, a, false, b,
                                                             (short)0, acc, false, false);
            }
            for (int r = 0; r < 8; ++r) {
                int Mrow = mt * 16 + r + hi8;
                if (Mrow < 25) {
                    float v = acc[r];
                    if (resid) v += (float)resid[Mrow][nt * 16 + nsel];
                    dst[Mrow][nt * 16 + nsel] = (_Float16)lrelu(v);
                }
            }
        }
    }
}

__global__ __launch_bounds__(256) void encoder_kernel(
    const float* __restrict__ obs,
    const float* b_c1, const float* b_r1a, const float* b_r1b,
    const float* b_r2a, const float* b_r2b, const float* b_c2, const float* b_lin,
    const _Float16* __restrict__ packW,
    float* __restrict__ encF, _Float16* __restrict__ encH) {

    __shared__ _Float16 obsT[8][50][32];  // 49 input positions + zero row, 7ch padded to 32
    __shared__ _Float16 xT[8][26][32];    // 25 positions + zero row
    __shared__ _Float16 hT[8][26][32];
    __shared__ _Float16 yv[9][416];       // flattened 400-vec (k=c*25+p), row 8 = zero row

    const int wave = threadIdx.x >> 5;
    const int lane = threadIdx.x & 31;
    const int agent = blockIdx.x * 8 + wave;
    const int klo = (lane >> 4) * 8;
    const int nsel = lane & 15;
    const int msel = lane & 15;
    const int hi8 = (lane >> 4) << 3;

    // --- init LDS ---
    for (int i = lane; i < 50 * 32; i += 32) ((_Float16*)obsT[wave])[i] = (_Float16)0;
    xT[wave][25][lane] = (_Float16)0;
    hT[wave][25][lane] = (_Float16)0;
    for (int i = threadIdx.x; i < 416; i += 256) yv[8][i] = (_Float16)0;
    for (int i = threadIdx.x; i < 8 * 16; i += 256) yv[i >> 4][400 + (i & 15)] = (_Float16)0;

    // --- load obs (f32 -> f16, [pos][ch]) ---
    const float* op = obs + (size_t)agent * 343;
    for (int idx = lane; idx < 343; idx += 32) {
        int c = idx / 49, r = idx % 49;
        obsT[wave][r][c] = (_Float16)op[idx];
    }
    __syncthreads();

    // --- conv1: VALID 3x3, 7(->32)ch -> 32ch, 9 taps of K=32 ---
    for (int nt = 0; nt < 2; ++nt) {
        float bv = b_c1[nt * 16 + nsel];
        for (int mt = 0; mt < 2; ++mt) {
            v8f acc = {bv, bv, bv, bv, bv, bv, bv, bv};
            int p = mt * 16 + msel;
            int py = p / 5, px = p % 5;
            for (int tap = 0; tap < 9; ++tap) {
                int ky = tap / 3, kx = tap % 3;
                int rin = (p < 25) ? ((py + ky) * 7 + (px + kx)) : 49;  // row 49 = zero row
                v16h a = load_a(&obsT[wave][rin][0], klo);
                v16h b = load_b(packW, C1T + tap * 2 + nt, lane);
                acc = __builtin_amdgcn_wmma_f32_16x16x32_f16(false, a, false, b,
                                                             (short)0, acc, false, false);
            }
            for (int r = 0; r < 8; ++r) {
                int Mrow = mt * 16 + r + hi8;
                if (Mrow < 25) xT[wave][Mrow][nt * 16 + nsel] = (_Float16)lrelu(acc[r]);
            }
        }
    }

    // --- residual blocks ---
    conv3x3_same(xT[wave], hT[wave], packW, R1AT, b_r1a, nullptr,  lane);
    conv3x3_same(hT[wave], xT[wave], packW, R1BT, b_r1b, xT[wave], lane);
    conv3x3_same(xT[wave], hT[wave], packW, R2AT, b_r2a, nullptr,  lane);
    conv3x3_same(hT[wave], xT[wave], packW, R2BT, b_r2b, xT[wave], lane);

    // --- 1x1 conv 32 -> 16, write flattened y (k = c*25 + p) ---
    {
        float bv = b_c2[nsel];
        for (int mt = 0; mt < 2; ++mt) {
            v8f acc = {bv, bv, bv, bv, bv, bv, bv, bv};
            int p = mt * 16 + msel;
            int rin = (p < 25) ? p : 25;
            v16h a = load_a(&xT[wave][rin][0], klo);
            v16h b = load_b(packW, C2T, lane);
            acc = __builtin_amdgcn_wmma_f32_16x16x32_f16(false, a, false, b,
                                                         (short)0, acc, false, false);
            for (int r = 0; r < 8; ++r) {
                int Mrow = mt * 16 + r + hi8;
                if (Mrow < 25) yv[wave][nsel * 25 + Mrow] = (_Float16)lrelu(acc[r]);
            }
        }
    }
    __syncthreads();

    // --- Linear 400(->416) -> 32 over the 8 agents of this WG (M-tile of 16, 8 pad rows) ---
    if (wave < 2) {
        int nt = wave;
        float bv = b_lin[nt * 16 + nsel];
        v8f acc = {bv, bv, bv, bv, bv, bv, bv, bv};
        for (int kt = 0; kt < 13; ++kt) {
            int row = (msel < 8) ? msel : 8;        // rows 8..15 -> zero row
            v16h a = load_a(&yv[row][kt * 32], klo);
            v16h b = load_b(packW, LINT + kt * 2 + nt, lane);
            acc = __builtin_amdgcn_wmma_f32_16x16x32_f16(false, a, false, b,
                                                         (short)0, acc, false, false);
        }
        for (int r = 0; r < 8; ++r) {
            int Mrow = r + hi8;
            if (Mrow < 8) {
                size_t ag = (size_t)blockIdx.x * 8 + Mrow;
                int n = nt * 16 + nsel;
                float v = acc[r];
                encF[ag * 32 + n] = v;
                encH[ag * 32 + n] = (_Float16)v;
            }
        }
    }
}

// ---------------------------------------------------------------------------
// Kernel 3: pair MLP. One wave per 16-pair tile; 8 waves per block.
// ---------------------------------------------------------------------------
__global__ __launch_bounds__(256) void pair_kernel(
    const int* __restrict__ pairs,
    const float* b_q1, const float* b_q2, const float* b_q3,
    const _Float16* __restrict__ packW,
    const float* __restrict__ encF, const _Float16* __restrict__ encH,
    float* __restrict__ outPE, float* __restrict__ outQ) {

    __shared__ int      pi[8][2][16];
    __shared__ _Float16 h1[8][16][32];
    __shared__ _Float16 h2[8][16][32];

    const int wave = threadIdx.x >> 5;
    const int lane = threadIdx.x & 31;
    const int tile = blockIdx.x * 8 + wave;
    const bool active = tile < (N_PAIRS / 16);
    const int base = tile * 16;
    const int klo = (lane >> 4) * 8;
    const int nsel = lane & 15;
    const int msel = lane & 15;
    const int hi8 = (lane >> 4) << 3;

    if (active && lane < 16) {
        pi[wave][0][lane] = pairs[(size_t)(base + lane) * 2 + 0];
        pi[wave][1][lane] = pairs[(size_t)(base + lane) * 2 + 1];
    }
    __syncthreads();

    if (active) {
        // stream pair_enc (f32 copy of gathered enc rows): lane side s, row (lane&15)
        {
            int row = lane & 15, s = lane >> 4;
            int idx = pi[wave][s][row];
            const float* src = encF + (size_t)idx * 32;
            float* dst = outPE + (size_t)(base + row) * 64 + s * 32;
            for (int i = 0; i < 32; i += 4)
                *(float4*)(dst + i) = *(const float4*)(src + i);
        }
        // q1: [16 x 64] @ [64 x 32], K-tiles select enc[i0] / enc[i1]
        {
            float bv0 = b_q1[nsel], bv1 = b_q1[16 + nsel];
            v8f a0 = {bv0, bv0, bv0, bv0, bv0, bv0, bv0, bv0};
            v8f a1 = {bv1, bv1, bv1, bv1, bv1, bv1, bv1, bv1};
            for (int kt = 0; kt < 2; ++kt) {
                int idx = pi[wave][kt][msel];
                v16h a = load_a(encH + (size_t)idx * 32, klo);
                v16h bA = load_b(packW, Q1T + kt * 2 + 0, lane);
                v16h bB = load_b(packW, Q1T + kt * 2 + 1, lane);
                a0 = __builtin_amdgcn_wmma_f32_16x16x32_f16(false, a, false, bA, (short)0, a0, false, false);
                a1 = __builtin_amdgcn_wmma_f32_16x16x32_f16(false, a, false, bB, (short)0, a1, false, false);
            }
            for (int r = 0; r < 8; ++r) {
                int M = r + hi8;
                h1[wave][M][nsel]      = (_Float16)lrelu(a0[r]);
                h1[wave][M][16 + nsel] = (_Float16)lrelu(a1[r]);
            }
        }
        // q2: [16 x 32] @ [32 x 32]
        {
            float bv0 = b_q2[nsel], bv1 = b_q2[16 + nsel];
            v8f a0 = {bv0, bv0, bv0, bv0, bv0, bv0, bv0, bv0};
            v8f a1 = {bv1, bv1, bv1, bv1, bv1, bv1, bv1, bv1};
            v16h a = load_a(&h1[wave][msel][0], klo);
            v16h bA = load_b(packW, Q2T + 0, lane);
            v16h bB = load_b(packW, Q2T + 1, lane);
            a0 = __builtin_amdgcn_wmma_f32_16x16x32_f16(false, a, false, bA, (short)0, a0, false, false);
            a1 = __builtin_amdgcn_wmma_f32_16x16x32_f16(false, a, false, bB, (short)0, a1, false, false);
            for (int r = 0; r < 8; ++r) {
                int M = r + hi8;
                h2[wave][M][nsel]      = (_Float16)lrelu(a0[r]);
                h2[wave][M][16 + nsel] = (_Float16)lrelu(a1[r]);
            }
        }
        // q3: [16 x 32] @ [32 x 2(pad 16)]
        {
            float bv = (nsel < 2) ? b_q3[nsel] : 0.f;
            v8f acc = {bv, bv, bv, bv, bv, bv, bv, bv};
            v16h a = load_a(&h2[wave][msel][0], klo);
            v16h b = load_b(packW, Q3T, lane);
            acc = __builtin_amdgcn_wmma_f32_16x16x32_f16(false, a, false, b, (short)0, acc, false, false);
            if (nsel < 2) {
                for (int r = 0; r < 8; ++r) {
                    int M = r + hi8;
                    outQ[(size_t)(base + M) * 2 + nsel] = acc[r];
                }
            }
        }
    }
}

// ---------------------------------------------------------------------------
extern "C" void kernel_launch(void* const* d_in, const int* in_sizes, int n_in,
                              void* d_out, int out_size, void* d_ws, size_t ws_size,
                              hipStream_t stream) {
    const float* obs   = (const float*)d_in[0];
    const int*   pairs = (const int*)d_in[1];
    const float* w_c1  = (const float*)d_in[2];  const float* b_c1  = (const float*)d_in[3];
    const float* w_r1a = (const float*)d_in[4];  const float* b_r1a = (const float*)d_in[5];
    const float* w_r1b = (const float*)d_in[6];  const float* b_r1b = (const float*)d_in[7];
    const float* w_r2a = (const float*)d_in[8];  const float* b_r2a = (const float*)d_in[9];
    const float* w_r2b = (const float*)d_in[10]; const float* b_r2b = (const float*)d_in[11];
    const float* w_c2  = (const float*)d_in[12]; const float* b_c2  = (const float*)d_in[13];
    const float* w_lin = (const float*)d_in[14]; const float* b_lin = (const float*)d_in[15];
    const float* w_q1  = (const float*)d_in[16]; const float* b_q1  = (const float*)d_in[17];
    const float* w_q2  = (const float*)d_in[18]; const float* b_q2  = (const float*)d_in[19];
    const float* w_q3  = (const float*)d_in[20]; const float* b_q3  = (const float*)d_in[21];

    _Float16* packW = (_Float16*)d_ws;                                        // 124 KB packed B tiles
    float*    encF  = (float*)((char*)d_ws + (128 << 10));                    // 8 MB enc f32
    _Float16* encH  = (_Float16*)((char*)d_ws + (128 << 10) + (8 << 20));     // 4 MB enc f16

    float* outPE = (float*)d_out;                       // [1M, 64]
    float* outQ  = outPE + (size_t)N_PAIRS * 64;        // [1M, 2]

    pack_weights<<<NTILES, 512, 0, stream>>>(w_c1, w_r1a, w_r1b, w_r2a, w_r2b,
                                             w_c2, w_lin, w_q1, w_q2, w_q3, packW);

    encoder_kernel<<<N_AGENTS / 8, 256, 0, stream>>>(obs, b_c1, b_r1a, b_r1b,
                                                     b_r2a, b_r2b, b_c2, b_lin,
                                                     packW, encF, encH);

    int ntile = N_PAIRS / 16;                // 62500 tiles, 8 per block
    pair_kernel<<<(ntile + 7) / 8, 256, 0, stream>>>(pairs, b_q1, b_q2, b_q3,
                                                     packW, encF, encH, outPE, outQ);
}
